// PointTransformerMat_50972671869350
// MI455X (gfx1250) — matrix operand order repaired and checked by reading, not compiled, not run on hardware
//
#include <hip/hip_runtime.h>
#include <hip/hip_bf16.h>
#include <cstdint>
#include <cstddef>

// ---------------------------------------------------------------------------
// CDNA5 (gfx1250, wave32) WMMA: v_wmma_f32_16x16x32_f16
// Fragment layouts per ISA 7.12.2.  LDS tiles are staged in "fragment-ready"
// layout: each lane's 16 half-elements are contiguous, so a fragment load is
// two ds_load_b128 and staging writes are single ds_store_b128.
//   A (16x32, MxK):  lane = (m&15) + 16*((k>>3)&1),  j = 8*(k>>4) + (k&7)
//   B (32x16, KxN):  lane = (n&15) + 16*((k>>3)&1),  j = 8*(k>>4) + (k&7)
// ---------------------------------------------------------------------------
typedef __attribute__((ext_vector_type(16))) _Float16 v16h;
typedef __attribute__((ext_vector_type(8)))  _Float16 f16x8;
typedef __attribute__((ext_vector_type(2)))  _Float16 f16x2;
typedef __attribute__((ext_vector_type(2)))  __fp16   fp16x2_raw;
typedef __attribute__((ext_vector_type(8)))  float    v8f;

#define WMMA_F16(a, b, c) \
  __builtin_amdgcn_wmma_f32_16x16x32_f16(false, (a), false, (b), (short)0, (c), false, false)

__device__ __forceinline__ f16x2 f2h2(float a, float b) {
#if __has_builtin(__builtin_amdgcn_cvt_pkrtz)
  const fp16x2_raw r = __builtin_amdgcn_cvt_pkrtz(a, b);  // v_cvt_pkrtz_f16_f32
  return __builtin_bit_cast(f16x2, r);
#else
  f16x2 r; r[0] = (_Float16)a; r[1] = (_Float16)b; return r;
#endif
}

// A fragment straight from global f32 (row-major, lda in elements), b128 loads.
// Requires 16B-aligned rows (lda % 4 == 0 and 16B-aligned base).
__device__ __forceinline__ v16h frag_a_f32v(const float* __restrict__ A, int lda) {
  const int lane = threadIdx.x & 31;
  const int m = lane & 15, g = lane >> 4;
  const float* p = A + (size_t)m * lda + 8 * g;
  const float4 f0 = *(const float4*)(p);
  const float4 f1 = *(const float4*)(p + 4);
  const float4 f2 = *(const float4*)(p + 16);
  const float4 f3 = *(const float4*)(p + 20);
  v16h o; f16x2* hp = (f16x2*)&o;
  hp[0] = f2h2(f0.x, f0.y); hp[1] = f2h2(f0.z, f0.w);
  hp[2] = f2h2(f1.x, f1.y); hp[3] = f2h2(f1.z, f1.w);
  hp[4] = f2h2(f2.x, f2.y); hp[5] = f2h2(f2.z, f2.w);
  hp[6] = f2h2(f3.x, f3.y); hp[7] = f2h2(f3.z, f3.w);
  return o;
}

// B fragment from an (N,K) row-major matrix used transposed (B[k][n]=S[n*ld+k]).
__device__ __forceinline__ v16h frag_bt_f32v(const float* __restrict__ S, int ld) {
  const int lane = threadIdx.x & 31;
  const int n = lane & 15, g = lane >> 4;
  const float* p = S + (size_t)n * ld + 8 * g;
  const float4 f0 = *(const float4*)(p);
  const float4 f1 = *(const float4*)(p + 4);
  const float4 f2 = *(const float4*)(p + 16);
  const float4 f3 = *(const float4*)(p + 20);
  v16h o; f16x2* hp = (f16x2*)&o;
  hp[0] = f2h2(f0.x, f0.y); hp[1] = f2h2(f0.z, f0.w);
  hp[2] = f2h2(f1.x, f1.y); hp[3] = f2h2(f1.z, f1.w);
  hp[4] = f2h2(f2.x, f2.y); hp[5] = f2h2(f2.z, f2.w);
  hp[6] = f2h2(f3.x, f3.y); hp[7] = f2h2(f3.z, f3.w);
  return o;
}

// B fragment, K-major strided source (B[k][n] = S[k*ldb + n]).
__device__ __forceinline__ v16h frag_b_f32(const float* __restrict__ B, int ldb) {
  const int lane = threadIdx.x & 31;
  const int n = lane & 15, g = lane >> 4;
  v16h out;
#pragma unroll
  for (int j = 0; j < 16; ++j) out[j] = (_Float16)B[(size_t)(g * 8 + (j & 7) + 16 * (j >> 3)) * ldb + n];
  return out;
}

// ---------------------------------------------------------------------------
// Fused GEMM: Y = act( bn( X*W^T + bias + res ) ), M assumed multiple of 64
// (true for every layer here: M = 16384 or 65536).
// Tile 64(M) x 128(N), 256 threads = 8 waves, each wave owns 32x32 out
// (4 wmma / 32-deep K-step).  Interior tiles take a branch-free b128 path;
// edge tiles use branchless clamped loads (no exec-mask serialization).
// ---------------------------------------------------------------------------
__global__ __launch_bounds__(256) void gemm_wmma_kernel(
    const float* __restrict__ X, int ldx,
    const float* __restrict__ W,
    const float* __restrict__ bias,
    const float* __restrict__ res, int ldres,
    const float* __restrict__ bng, const float* __restrict__ bnb,
    const float* __restrict__ bnm, const float* __restrict__ bnv,
    float* __restrict__ Y, int ldy,
    int M, int N, int K, int act) {
  __shared__ __align__(32) _Float16 As[4][32][16];   // [mtile][lane][j]  4KB
  __shared__ __align__(32) _Float16 Bs[8][32][16];   // [ntile][lane][j]  8KB
  const int n0 = blockIdx.x * 128, m0 = blockIdx.y * 64;
  const int t = threadIdx.x;
  const int w = t >> 5, lane = t & 31;
  const int wr = w & 1, wc = w >> 1;                 // wave: rows wr*32, cols wc*32
  const bool vecA = ((ldx & 3) == 0);
  const bool vecB = ((K & 3) == 0);
  v8f acc[2][2] = {};
  const int ksteps = (K + 31) >> 5;
  for (int kt = 0; kt < ksteps; ++kt) {
    const int k0 = kt * 32;
    const bool fullK = (k0 + 32 <= K);
    const bool fullN = (n0 + 128 <= N);
    // ---- stage A (64x32) ----
    if (fullK && vecA) {
      const int mm = t >> 2, kb = t & 3;             // 256 tasks, 8 k each
      const float* Xp = X + (size_t)(m0 + mm) * ldx + k0 + kb * 8;
      if (k0 + 32 < K) __builtin_prefetch(Xp + 32, 0, 1);
      const float4 f0 = *(const float4*)(Xp);
      const float4 f1 = *(const float4*)(Xp + 4);
      f16x8 hv; f16x2* hp = (f16x2*)&hv;
      hp[0] = f2h2(f0.x, f0.y); hp[1] = f2h2(f0.z, f0.w);
      hp[2] = f2h2(f1.x, f1.y); hp[3] = f2h2(f1.z, f1.w);
      *(f16x8*)&As[mm >> 4][(mm & 15) + 16 * (kb & 1)][8 * (kb >> 1)] = hv;
    } else {
#pragma unroll
      for (int i = 0; i < 8; ++i) {                  // branchless clamped loads
        const int e = i * 256 + t;
        const int mm = e >> 5, kk = e & 31;
        const int kc = (k0 + kk < K) ? (k0 + kk) : (K - 1);
        float vx = X[(size_t)(m0 + mm) * ldx + kc];
        if (k0 + kk >= K) vx = 0.f;
        As[mm >> 4][(mm & 15) + 16 * ((kk >> 3) & 1)][8 * (kk >> 4) + (kk & 7)] = (_Float16)vx;
      }
    }
    // ---- stage B (32x128) = W^T slice ----
    if (fullK && fullN && vecB) {
#pragma unroll
      for (int u = t; u < 512; u += 256) {           // 512 tasks, 8 k each
        const int nn = u >> 2, kb = u & 3;
        const float* Wp = W + (size_t)(n0 + nn) * K + k0 + kb * 8;
        const float4 f0 = *(const float4*)(Wp);
        const float4 f1 = *(const float4*)(Wp + 4);
        f16x8 hv; f16x2* hp = (f16x2*)&hv;
        hp[0] = f2h2(f0.x, f0.y); hp[1] = f2h2(f0.z, f0.w);
        hp[2] = f2h2(f1.x, f1.y); hp[3] = f2h2(f1.z, f1.w);
        *(f16x8*)&Bs[nn >> 4][(nn & 15) + 16 * (kb & 1)][8 * (kb >> 1)] = hv;
      }
    } else {
#pragma unroll
      for (int i = 0; i < 16; ++i) {
        const int e = i * 256 + t;
        const int nn = e >> 5, kk = e & 31;
        const int nc = (n0 + nn < N) ? (n0 + nn) : (N - 1);
        const int kc = (k0 + kk < K) ? (k0 + kk) : (K - 1);
        float vw = W[(size_t)nc * K + kc];
        if (n0 + nn >= N || k0 + kk >= K) vw = 0.f;
        Bs[nn >> 4][(nn & 15) + 16 * ((kk >> 3) & 1)][8 * (kk >> 4) + (kk & 7)] = (_Float16)vw;
      }
    }
    __syncthreads();
    const v16h a0 = *(const v16h*)&As[wr * 2 + 0][lane][0];
    const v16h a1 = *(const v16h*)&As[wr * 2 + 1][lane][0];
    const v16h b0 = *(const v16h*)&Bs[wc * 2 + 0][lane][0];
    const v16h b1 = *(const v16h*)&Bs[wc * 2 + 1][lane][0];
    acc[0][0] = WMMA_F16(a0, b0, acc[0][0]);
    acc[0][1] = WMMA_F16(a0, b1, acc[0][1]);
    acc[1][0] = WMMA_F16(a1, b0, acc[1][0]);
    acc[1][1] = WMMA_F16(a1, b1, acc[1][1]);
    __syncthreads();
  }
  // fused epilogue: bias + residual + folded-BN + activation
  const int gg = lane >> 4, nl = lane & 15;
#pragma unroll
  for (int ni = 0; ni < 2; ++ni) {
    const int ncol = n0 + wc * 32 + ni * 16 + nl;
    if (ncol < N) {
      const float bv = bias ? bias[ncol] : 0.f;
      float scale = 1.f, shift = 0.f;
      if (bng) {
        const float inv = rsqrtf(bnv[ncol] + 1e-5f);
        scale = bng[ncol] * inv;
        shift = bnb[ncol] - bnm[ncol] * scale;
      }
#pragma unroll
      for (int mi = 0; mi < 2; ++mi) {
#pragma unroll
        for (int r = 0; r < 8; ++r) {
          const int mm = m0 + wr * 32 + mi * 16 + 8 * gg + r;
          float v = acc[mi][ni][r] + bv;
          if (res) v += res[(size_t)mm * ldres + ncol];
          if (bng) v = v * scale + shift;
          if (act == 1)      v = v > 0.f ? v : 0.f;
          else if (act == 2) v = v > 0.f ? v : 0.2f * v;
          else if (act == 3) v = 1.f / (1.f + __expf(-v));
          Y[(size_t)mm * ldy + ncol] = v;
        }
      }
    }
  }
}

// ---------------------------------------------------------------------------
// Per-group MHA: block = 1 group (32 samples x 128ch), wave = 1 head (d=32).
// QK^T and P*V are 2x2 tiles of v_wmma_f32_16x16x32_f16.  Probabilities are
// written to LDS in fragment-ready layout (two ds_store_b128 per row).
// ---------------------------------------------------------------------------
__global__ __launch_bounds__(128) void attn_kernel(const float* __restrict__ qkv,
                                                   float* __restrict__ out) {
  __shared__ __align__(16) float    sc[4][32][36];       // scores (16B stride)
  __shared__ __align__(32) _Float16 paf[4][2][32][16];   // probs, frag-ready
  const int gid = blockIdx.x;
  const int h = threadIdx.x >> 5;
  const int lane = threadIdx.x & 31;
  const float* q  = qkv + (size_t)gid * 32 * 384 + h * 32;
  const float* km = q + 128;
  const float* vm = q + 256;
  const float scale = 0.17677669529663687f;  // 1/sqrt(32)

  const v16h a0  = frag_a_f32v(q, 384);
  const v16h a1  = frag_a_f32v(q + 16 * 384, 384);
  const v16h bk0 = frag_bt_f32v(km, 384);
  const v16h bk1 = frag_bt_f32v(km + 16 * 384, 384);
  v8f s00 = {}, s01 = {}, s10 = {}, s11 = {};
  s00 = WMMA_F16(a0, bk0, s00);
  s01 = WMMA_F16(a0, bk1, s01);
  s10 = WMMA_F16(a1, bk0, s10);
  s11 = WMMA_F16(a1, bk1, s11);

  const int gg = lane >> 4, nl = lane & 15;
#pragma unroll
  for (int r = 0; r < 8; ++r) {
    sc[h][8 * gg + r][nl]           = s00[r] * scale;
    sc[h][8 * gg + r][16 + nl]      = s01[r] * scale;
    sc[h][16 + 8 * gg + r][nl]      = s10[r] * scale;
    sc[h][16 + 8 * gg + r][16 + nl] = s11[r] * scale;
  }
  // softmax: one row per lane, b128 LDS reads
  float pr[32];
#pragma unroll
  for (int j4 = 0; j4 < 8; ++j4) {
    const float4 v = *(const float4*)&sc[h][lane][j4 * 4];
    pr[4 * j4] = v.x; pr[4 * j4 + 1] = v.y; pr[4 * j4 + 2] = v.z; pr[4 * j4 + 3] = v.w;
  }
  float mx = -3.4e38f;
#pragma unroll
  for (int j = 0; j < 32; ++j) mx = fmaxf(mx, pr[j]);
  float sum = 0.f;
#pragma unroll
  for (int j = 0; j < 32; ++j) { pr[j] = __expf(pr[j] - mx); sum += pr[j]; }
  const float inv = 1.f / sum;
#pragma unroll
  for (int j = 0; j < 32; ++j) pr[j] *= inv;
  // fragment-ready store: lane slot ml holds k{0..7,16..23}, ml+16 k{8..15,24..31}
  {
    const int mt = lane >> 4, ml = lane & 15;
    v16h lo, hi; f16x2* lp = (f16x2*)&lo; f16x2* hp = (f16x2*)&hi;
    lp[0] = f2h2(pr[0],  pr[1]);  lp[1] = f2h2(pr[2],  pr[3]);
    lp[2] = f2h2(pr[4],  pr[5]);  lp[3] = f2h2(pr[6],  pr[7]);
    lp[4] = f2h2(pr[16], pr[17]); lp[5] = f2h2(pr[18], pr[19]);
    lp[6] = f2h2(pr[20], pr[21]); lp[7] = f2h2(pr[22], pr[23]);
    hp[0] = f2h2(pr[8],  pr[9]);  hp[1] = f2h2(pr[10], pr[11]);
    hp[2] = f2h2(pr[12], pr[13]); hp[3] = f2h2(pr[14], pr[15]);
    hp[4] = f2h2(pr[24], pr[25]); hp[5] = f2h2(pr[26], pr[27]);
    hp[6] = f2h2(pr[28], pr[29]); hp[7] = f2h2(pr[30], pr[31]);
    *(v16h*)&paf[h][mt][ml][0]      = lo;
    *(v16h*)&paf[h][mt][ml + 16][0] = hi;
  }
  const v16h p0  = *(const v16h*)&paf[h][0][lane][0];
  const v16h p1  = *(const v16h*)&paf[h][1][lane][0];
  const v16h bv0 = frag_b_f32(vm, 384);
  const v16h bv1 = frag_b_f32(vm + 16, 384);
  v8f o00 = {}, o01 = {}, o10 = {}, o11 = {};
  o00 = WMMA_F16(p0, bv0, o00);
  o01 = WMMA_F16(p0, bv1, o01);
  o10 = WMMA_F16(p1, bv0, o10);
  o11 = WMMA_F16(p1, bv1, o11);

  float* ob = out + (size_t)gid * 32 * 128 + h * 32;
#pragma unroll
  for (int r = 0; r < 8; ++r) {
    ob[(size_t)(8 * gg + r) * 128 + nl]           = o00[r];
    ob[(size_t)(8 * gg + r) * 128 + 16 + nl]      = o01[r];
    ob[(size_t)(16 + 8 * gg + r) * 128 + nl]      = o10[r];
    ob[(size_t)(16 + 8 * gg + r) * 128 + 16 + nl] = o11[r];
  }
}

// ---------------------------------------------------------------------------
// Point-cloud normalization (per-batch mean / max-norm, LDS reductions)
// ---------------------------------------------------------------------------
__global__ __launch_bounds__(256) void normalize_pc_kernel(const float* __restrict__ xyz,
                                                           float* __restrict__ out, int Nn) {
  const int b = blockIdx.x;
  const float* P = xyz + (size_t)b * Nn * 3;
  float* O = out + (size_t)b * Nn * 3;
  __shared__ float rx[256], ry[256], rz[256];
  const int t = threadIdx.x;
  float sx = 0.f, sy = 0.f, sz = 0.f;
  for (int i = t; i < Nn; i += 256) { sx += P[i * 3]; sy += P[i * 3 + 1]; sz += P[i * 3 + 2]; }
  rx[t] = sx; ry[t] = sy; rz[t] = sz;
  __syncthreads();
  for (int s = 128; s > 0; s >>= 1) {
    if (t < s) { rx[t] += rx[t + s]; ry[t] += ry[t + s]; rz[t] += rz[t + s]; }
    __syncthreads();
  }
  const float mxm = rx[0] / Nn, mym = ry[0] / Nn, mzm = rz[0] / Nn;
  __syncthreads();
  float mx = 0.f;
  for (int i = t; i < Nn; i += 256) {
    const float dx = P[i * 3] - mxm, dy = P[i * 3 + 1] - mym, dz = P[i * 3 + 2] - mzm;
    mx = fmaxf(mx, sqrtf(dx * dx + dy * dy + dz * dz));
  }
  rx[t] = mx;
  __syncthreads();
  for (int s = 128; s > 0; s >>= 1) {
    if (t < s) rx[t] = fmaxf(rx[t], rx[t + s]);
    __syncthreads();
  }
  const float inv = 1.f / (rx[0] + 1e-8f);
  for (int i = t; i < Nn; i += 256) {
    O[i * 3]     = (P[i * 3] - mxm) * inv;
    O[i * 3 + 1] = (P[i * 3 + 1] - mym) * inv;
    O[i * 3 + 2] = (P[i * 3 + 2] - mzm) * inv;
  }
}

// ---------------------------------------------------------------------------
// Farthest point sampling: dmin stays LDS-resident (32KB of 320KB WGP LDS)
// ---------------------------------------------------------------------------
__global__ __launch_bounds__(256) void fps_kernel(const float* __restrict__ xyz,
                                                  int* __restrict__ fps_idx,
                                                  int Nn, int npoint) {
  const int b = blockIdx.x;
  const float* P = xyz + (size_t)b * Nn * 3;
  __shared__ float dmin[8192];
  __shared__ float rmax[256];
  __shared__ int   rarg[256];
  __shared__ int   curS;
  const int t = threadIdx.x;
  for (int i = t; i < Nn; i += 256) dmin[i] = 1e10f;
  if (t == 0) curS = 0;
  __syncthreads();
  for (int it = 0; it < npoint; ++it) {
    const int far = curS;
    if (t == 0) fps_idx[b * npoint + it] = far;
    const float cx = P[far * 3], cy = P[far * 3 + 1], cz = P[far * 3 + 2];
    float bmax = -1.f; int barg = 0;
    for (int i = t; i < Nn; i += 256) {
      const float dx = P[i * 3] - cx, dy = P[i * 3 + 1] - cy, dz = P[i * 3 + 2] - cz;
      float dm = fminf(dmin[i], dx * dx + dy * dy + dz * dz);
      dmin[i] = dm;
      if (dm > bmax) { bmax = dm; barg = i; }
    }
    rmax[t] = bmax; rarg[t] = barg;
    __syncthreads();
    for (int s = 128; s > 0; s >>= 1) {
      if (t < s && rmax[t + s] > rmax[t]) { rmax[t] = rmax[t + s]; rarg[t] = rarg[t + s]; }
      __syncthreads();
    }
    if (t == 0) curS = rarg[0];
    __syncthreads();
  }
}

__global__ void gather_new_xyz_kernel(const float* __restrict__ xyzn,
                                      const int* __restrict__ fpsi,
                                      float* __restrict__ nxyz,
                                      int Nn, int npoint, int total) {
  const int i = blockIdx.x * blockDim.x + threadIdx.x;
  if (i >= total) return;
  const int c = i % 3, j = i / 3;
  const int b = j / npoint, q = j % npoint;
  const int pi = fpsi[b * npoint + q];
  nxyz[i] = xyzn[((size_t)b * Nn + pi) * 3 + c];
}

// ---------------------------------------------------------------------------
// kNN: block = 1 query; 8192 distances in LDS; 32 argmin sweeps w/ tie-break
// ---------------------------------------------------------------------------
__global__ __launch_bounds__(256) void knn_kernel(const float* __restrict__ xyz,
                                                  const float* __restrict__ nxyz,
                                                  int* __restrict__ idx,
                                                  int Nn, int npoint, int nsample) {
  const int q = blockIdx.x, b = blockIdx.y;
  __shared__ float dist[8192];
  __shared__ float rmin[256];
  __shared__ int   rarg[256];
  const float* P = xyz + (size_t)b * Nn * 3;
  const float qx = nxyz[(size_t)(b * npoint + q) * 3];
  const float qy = nxyz[(size_t)(b * npoint + q) * 3 + 1];
  const float qz = nxyz[(size_t)(b * npoint + q) * 3 + 2];
  const int t = threadIdx.x;
  for (int i = t; i < Nn; i += 256) {
    const float dx = P[i * 3] - qx, dy = P[i * 3 + 1] - qy, dz = P[i * 3 + 2] - qz;
    dist[i] = dx * dx + dy * dy + dz * dz;
  }
  __syncthreads();
  for (int s = 0; s < nsample; ++s) {
    float bmin = 3.4e38f; int barg = 0x7fffffff;
    for (int i = t; i < Nn; i += 256) {
      const float d = dist[i];
      if (d < bmin || (d == bmin && i < barg)) { bmin = d; barg = i; }
    }
    rmin[t] = bmin; rarg[t] = barg;
    __syncthreads();
    for (int st = 128; st > 0; st >>= 1) {
      if (t < st) {
        if (rmin[t + st] < rmin[t] || (rmin[t + st] == rmin[t] && rarg[t + st] < rarg[t])) {
          rmin[t] = rmin[t + st]; rarg[t] = rarg[t + st];
        }
      }
      __syncthreads();
    }
    if (t == 0) {
      idx[((size_t)(b * npoint + q)) * nsample + s] = rarg[0];
      dist[rarg[0]] = 3.4e38f;
    }
    __syncthreads();
  }
}

__global__ __launch_bounds__(128) void gather_group_kernel(const float* __restrict__ points,
                                                           const int* __restrict__ idx,
                                                           float* __restrict__ out,
                                                           int Nn, int gpb) {
  const int j = blockIdx.x;
  const int b = j / gpb;
  const int pi = idx[j];
  const int c = threadIdx.x;
  out[(size_t)j * 128 + c] = points[((size_t)b * Nn + pi) * 128 + c];
}

__global__ __launch_bounds__(128) void scatter_kernel(const float* __restrict__ g,
                                                      const int* __restrict__ idx,
                                                      float* __restrict__ recon,
                                                      float* __restrict__ cnt,
                                                      int Nn, int gpb) {
  const int j = blockIdx.x;
  const int b = j / gpb;
  const int pi = idx[j];
  const int c = threadIdx.x;
  atomicAdd(&recon[((size_t)b * Nn + pi) * 128 + c], g[(size_t)j * 128 + c]);
  if (c == 0) atomicAdd(&cnt[(size_t)b * Nn + pi], 1.0f);
}

__global__ __launch_bounds__(256) void concat_kernel(const float* __restrict__ points,
                                                     const float* __restrict__ recon,
                                                     const float* __restrict__ cnt,
                                                     float* __restrict__ out) {
  const size_t i = blockIdx.x;
  const int t = threadIdx.x;
  float v;
  if (t < 128) v = points[i * 128 + t];
  else         v = recon[i * 128 + (t - 128)] / fmaxf(cnt[i], 1.0f);
  out[i * 256 + t] = v;
}

__global__ void xyz_to_points_kernel(const float* __restrict__ xyzn,
                                     float* __restrict__ points, int total) {
  const int i = blockIdx.x * blockDim.x + threadIdx.x;
  if (i >= total) return;
  const size_t p = i / 3;
  points[p * 128 + (i % 3)] = xyzn[i];
}

__global__ void fill_kernel(float* __restrict__ p, float v, size_t n) {
  const size_t i = (size_t)blockIdx.x * blockDim.x + threadIdx.x;
  if (i < n) p[i] = v;
}

__global__ void transpose_out_kernel(const float* __restrict__ y,
                                     float* __restrict__ out, int Nn, int total) {
  const int i = blockIdx.x * blockDim.x + threadIdx.x;
  if (i >= total) return;
  const int c = i % 5;
  const size_t m = i / 5;
  const int n = (int)(m % Nn), b = (int)(m / Nn);
  out[((size_t)b * 5 + c) * Nn + n] = y[i];
}

// ---------------------------------------------------------------------------
// Host-side orchestration
// ---------------------------------------------------------------------------
enum {
  IN_XYZ = 0, IN_FEATS = 1,
  E_L1W = 2, E_L1B, E_BN1G, E_BN1B, E_BN1M, E_BN1V,
  E_L2W, E_L2B, E_BN2G, E_BN2B, E_BN2M, E_BN2V,
  M_C1W, M_C1B, M_BN1G, M_BN1B, M_BN1M, M_BN1V,
  M_C2W, M_C2B, M_BN2G, M_BN2B, M_BN2M, M_BN2V,
  L_BASE = 26,   // per layer: in_w,in_b,out_w,out_b,bn_g,bn_b,bn_m,bn_v,lin_w,lin_b
  D_L1W = 66, D_L1B, D_BN1G, D_BN1B, D_BN1M, D_BN1V,
  D_L2W = 72, D_L2B, D_BN2G, D_BN2B, D_BN2M, D_BN2V,
  H_L1W = 78, H_L1B, H_BN1G, H_BN1B, H_BN1M, H_BN1V,
  H_L2W = 84, H_L2B, H_BN2G, H_BN2B, H_BN2M, H_BN2V,
  H_L3W = 90, H_L3B = 91
};

static inline void launch_gemm(hipStream_t s,
                               const float* X, int ldx, const float* W, const float* bias,
                               const float* res, int ldres,
                               const float* bng, const float* bnb,
                               const float* bnm, const float* bnv,
                               float* Y, int ldy, int M, int N, int K, int act) {
  dim3 grid((N + 127) / 128, (M + 63) / 64);
  gemm_wmma_kernel<<<grid, dim3(256), 0, s>>>(X, ldx, W, bias, res, ldres,
                                              bng, bnb, bnm, bnv, Y, ldy, M, N, K, act);
}

extern "C" void kernel_launch(void* const* d_in, const int* in_sizes, int n_in,
                              void* d_out, int out_size, void* d_ws, size_t ws_size,
                              hipStream_t stream) {
  (void)in_sizes; (void)n_in; (void)out_size; (void)ws_size;
  constexpr int Bb = 2, Nn = 8192, NPT = 1024, NS = 32;
  constexpr int BN = Bb * Nn;      // 16384 points
  constexpr int G  = Bb * NPT;     // 2048 groups
  constexpr int GS = G * NS;       // 65536 grouped samples

  auto F = [&](int i) { return (const float*)d_in[i]; };
  const float* xyz   = F(IN_XYZ);
  const float* feats = F(IN_FEATS);

  float* ws = (float*)d_ws;
  size_t off = 0;
  auto alloc = [&](size_t n) { float* p = ws + off; off += n; return p; };
  float* xyzn   = alloc((size_t)BN * 3);
  float* nxyz   = alloc((size_t)G * 3);
  float* points = alloc((size_t)BN * 128);
  float* gx     = alloc((size_t)GS * 128);   // also hosts h155 and head-512
  float* go     = alloc((size_t)GS * 128);   // also hosts head-256
  float* gt     = alloc((size_t)GS * 128);
  float* big    = alloc((size_t)BN * 2048);  // hosts qkv (GS*384) then d1
  float* recon  = alloc((size_t)BN * 128);
  float* cnt    = alloc((size_t)BN);
  float* cat    = alloc((size_t)BN * 256);
  float* d2     = alloc((size_t)BN * 1024);
  float* out5   = alloc((size_t)BN * 8);
  int*   fpsi   = (int*)alloc(G);
  int*   knni   = (int*)alloc(GS);

  // 1) normalize point cloud
  normalize_pc_kernel<<<Bb, 256, 0, stream>>>(xyz, xyzn, Nn);

  // 2) input embedding: 155 -> 155 (relu) -> 125 (relu) into points[:,3:]
  launch_gemm(stream, feats, 155, F(E_L1W), F(E_L1B), nullptr, 0,
              F(E_BN1G), F(E_BN1B), F(E_BN1M), F(E_BN1V), gx, 155, BN, 155, 155, 1);
  launch_gemm(stream, gx, 155, F(E_L2W), F(E_L2B), nullptr, 0,
              F(E_BN2G), F(E_BN2B), F(E_BN2M), F(E_BN2V), points + 3, 128, BN, 125, 155, 1);
  xyz_to_points_kernel<<<(BN * 3 + 255) / 256, 256, 0, stream>>>(xyzn, points, BN * 3);

  // 3) FPS + kNN grouping
  fps_kernel<<<Bb, 256, 0, stream>>>(xyzn, fpsi, Nn, NPT);
  gather_new_xyz_kernel<<<(G * 3 + 255) / 256, 256, 0, stream>>>(xyzn, fpsi, nxyz, Nn, NPT, G * 3);
  knn_kernel<<<dim3(NPT, Bb), 256, 0, stream>>>(xyzn, nxyz, knni, Nn, NPT, NS);
  gather_group_kernel<<<GS, 128, 0, stream>>>(points, knni, gx, Nn, NPT * NS);

  // 4) MSA: c1, c2
  launch_gemm(stream, gx, 128, F(M_C1W), F(M_C1B), nullptr, 0,
              F(M_BN1G), F(M_BN1B), F(M_BN1M), F(M_BN1V), gt, 128, GS, 128, 128, 1);
  launch_gemm(stream, gt, 128, F(M_C2W), F(M_C2B), nullptr, 0,
              F(M_BN2G), F(M_BN2B), F(M_BN2M), F(M_BN2V), gx, 128, GS, 128, 128, 1);

  // 5) 4 stacked attention layers
  for (int l = 0; l < 4; ++l) {
    const int base = L_BASE + 10 * l;
    launch_gemm(stream, gx, 128, F(base + 0), F(base + 1), nullptr, 0,
                nullptr, nullptr, nullptr, nullptr, big, 384, GS, 384, 128, 0);
    attn_kernel<<<G, 128, 0, stream>>>(big, go);
    launch_gemm(stream, go, 128, F(base + 2), F(base + 3), gx, 128,
                F(base + 4), F(base + 5), F(base + 6), F(base + 7), gt, 128, GS, 128, 128, 0);
    launch_gemm(stream, gt, 128, F(base + 8), F(base + 9), gt, 128,
                nullptr, nullptr, nullptr, nullptr, gx, 128, GS, 128, 128, 0);
  }

  // 6) scatter-add average reconstruction (re-zero every call: atomics)
  fill_kernel<<<((size_t)BN * 128 + 255) / 256, 256, 0, stream>>>(recon, 0.f, (size_t)BN * 128);
  fill_kernel<<<(BN + 255) / 256, 256, 0, stream>>>(cnt, 0.f, (size_t)BN);
  scatter_kernel<<<GS, 128, 0, stream>>>(gx, knni, recon, cnt, Nn, NPT * NS);
  concat_kernel<<<BN, 256, 0, stream>>>(points, recon, cnt, cat);

  // 7) down-projections (leaky 0.2)
  launch_gemm(stream, cat, 256, F(D_L1W), F(D_L1B), nullptr, 0,
              F(D_BN1G), F(D_BN1B), F(D_BN1M), F(D_BN1V), big, 2048, BN, 2048, 256, 2);
  launch_gemm(stream, big, 2048, F(D_L2W), F(D_L2B), nullptr, 0,
              F(D_BN2G), F(D_BN2B), F(D_BN2M), F(D_BN2V), d2, 1024, BN, 1024, 2048, 2);

  // 8) head
  launch_gemm(stream, d2, 1024, F(H_L1W), F(H_L1B), nullptr, 0,
              F(H_BN1G), F(H_BN1B), F(H_BN1M), F(H_BN1V), gx, 512, BN, 512, 1024, 1);
  launch_gemm(stream, gx, 512, F(H_L2W), F(H_L2B), nullptr, 0,
              F(H_BN2G), F(H_BN2B), F(H_BN2M), F(H_BN2V), go, 256, BN, 256, 512, 1);
  launch_gemm(stream, go, 256, F(H_L3W), F(H_L3B), nullptr, 0,
              nullptr, nullptr, nullptr, nullptr, out5, 5, BN, 5, 256, 3);

  // 9) (B,N,5) -> (B,5,N)
  transpose_out_kernel<<<(BN * 5 + 255) / 256, 256, 0, stream>>>(out5, (float*)d_out, Nn, BN * 5);
}
